// ColBERT_15118284882564
// MI455X (gfx1250) — matrix operand order repaired and checked by reading, compiled, not run on hardware
//
#include <hip/hip_runtime.h>
#include <hip/hip_bf16.h>
#include <stdint.h>

// ---------------------------------------------------------------------------
// ColBERT scoring for MI455X (gfx1250, wave32, WMMA + async-to-LDS staging).
//   qe = l2norm(relu(Q@W1+b1)@W2+b2 * qmask)   [64,32,128]
//   de = l2norm(relu(D@W1+b1)@W2+b2 * dmask)   [256,180,128]
//   out[q,d] = sum_i max_j (qe[q,i]·de[d,j]) / qlen[q]
// GEMMs: v_wmma_f32_16x16x32_bf16 (bf16 in, f32 accum).
// Hidden-state tiles staged via global_load_async_to_lds_b128 (ASYNCcnt).
// ---------------------------------------------------------------------------

typedef __bf16 bf16_t;
typedef __attribute__((ext_vector_type(16))) __bf16 v16bf;
typedef __attribute__((ext_vector_type(8)))  __bf16 v8bf;
typedef __attribute__((ext_vector_type(8)))  float  v8f;

#define HDIM 768
#define EDIM 128
#define NQ 64
#define LQ 32
#define ND 256
#define LD 180

// ---- WMMA helpers ---------------------------------------------------------

static __device__ __forceinline__ v8f wmma_bf16(v16bf a, v16bf b, v8f c) {
  return __builtin_amdgcn_wmma_f32_16x16x32_bf16(
      /*neg_a=*/false, a, /*neg_b=*/false, b,
      /*c_mod=*/(short)0, c, /*reuse_a=*/false, /*reuse_b=*/false);
}

// A-matrix 16x32 bf16 fragment: lane l holds row r=l&15;
// half h=l>>4 selects contiguous K range {h*16..h*16+15} (matches B loads).
static __device__ __forceinline__ v16bf a_frag(const bf16_t* row, int k0, int h) {
  v16bf a;
  ((v8bf*)&a)[0] = *(const v8bf*)(row + k0 + h * 16);
  ((v8bf*)&a)[1] = *(const v8bf*)(row + k0 + h * 16 + 8);
  return a;
}

static __device__ __forceinline__ v16bf zero16() {
  v16bf z;
#pragma unroll
  for (int i = 0; i < 16; ++i) z[i] = (bf16_t)0.0f;
  return z;
}

// ---- Kernel 1: weight fp32 -> bf16 transpose ------------------------------
// W1t[n][k] = W1[k][n]  (768x768) ; W2t[e][k] = W2[k][e]  (128x768)
__global__ void convert_weights(const float* __restrict__ W1,
                                const float* __restrict__ W2,
                                bf16_t* __restrict__ W1t,
                                bf16_t* __restrict__ W2t) {
  int idx = blockIdx.x * blockDim.x + threadIdx.x;
  const int total1 = HDIM * HDIM;
  if (idx < total1) {
    int k = idx / HDIM, n = idx % HDIM;            // coalesced read
    W1t[(size_t)n * HDIM + k] = (bf16_t)W1[idx];
  } else {
    int i2 = idx - total1;
    if (i2 < HDIM * EDIM) {
      int k = i2 / EDIM, e = i2 % EDIM;
      W2t[(size_t)e * HDIM + k] = (bf16_t)W2[i2];
    }
  }
}

// ---- Kernel 2: fused projection head --------------------------------------
// One block (256 thr = 8 waves) per 32 tokens (2 WMMA M-tiles).
//   T = relu(X @ W1 + b1)  (32x768, LDS bf16); Y = T @ W2 + b2  (32x128)
//   out = (Y * mask_row) / max(||Y*mask||, 1e-12)   stored bf16
// Layer 1: wave owns 6 N-tiles, B fragment shared by both M-tiles (halves
// W1t L2 traffic per token vs 16-token blocks). Layer 2: wave owns 1 N-tile.
__global__ __launch_bounds__(256) void proj_kernel(
    const float* __restrict__ hidden,   // [ntok, 768] fp32
    const int*   __restrict__ mask,     // [ntok]
    const bf16_t* __restrict__ W1t,     // [768 out][768 in] bf16
    const float* __restrict__ b1,       // [768]
    const bf16_t* __restrict__ W2t,     // [128 out][768 in] bf16
    const float* __restrict__ b2,       // [128]
    bf16_t* __restrict__ out)           // [ntok, 128] bf16
{
  // 144 KB LDS overlay: Xstage (96K f32, dead after convert) / Ts aliases it.
  __shared__ __align__(16) char smem_[147456];
  float*  Xstage = (float*)smem_;             // [32][768] f32
  bf16_t* Ts     = (bf16_t*)smem_;            // [32][768] bf16 (aliases Xstage)
  bf16_t* Xs     = (bf16_t*)(smem_ + 98304);  // [32][768] bf16
  __shared__ float rowsq[32];

  const int tid  = threadIdx.x;
  const int wave = tid >> 5;
  const int lane = tid & 31;
  const int h    = lane >> 4;           // lane half (A/B K-group select)
  const int n    = lane & 15;           // B/C/D column within tile; also A row
  const long tb  = (long)blockIdx.x * 32;

  // ---- Stage 32x768 fp32 tile into LDS with the async-to-LDS engine ----
  const float* src = hidden + (size_t)tb * HDIM;
  for (int g = tid; g < 32 * HDIM / 4; g += 256) {
    unsigned lds = (unsigned)(uintptr_t)&Xstage[g * 4];
    unsigned long long ga = (unsigned long long)(src + (size_t)g * 4);
    asm volatile("global_load_async_to_lds_b128 %0, %1, off"
                 :: "v"(lds), "v"(ga) : "memory");
  }
  asm volatile("s_wait_asynccnt 0" ::: "memory");
  __syncthreads();
  // Convert staged f32 -> bf16 (Xs).
  for (int g = tid; g < 32 * HDIM / 4; g += 256) {
    float4 x = ((const float4*)Xstage)[g];
    bf16_t* p = &Xs[g * 4];
    p[0] = (bf16_t)x.x; p[1] = (bf16_t)x.y;
    p[2] = (bf16_t)x.z; p[3] = (bf16_t)x.w;
  }
  __syncthreads();   // Xstage dead from here; Ts may overwrite it

  // ---- Layer 1: each wave owns 6 N-tiles x 2 M-tiles ----
  const int nbase = wave * 6;
  v8f acc[2][6];
#pragma unroll
  for (int t = 0; t < 6; ++t) {
    const float bv = b1[(nbase + t) * 16 + n];
#pragma unroll
    for (int v = 0; v < 8; ++v) { acc[0][t][v] = bv; acc[1][t][v] = bv; }
  }
  const bf16_t* arow0 = &Xs[n * HDIM];
  const bf16_t* arow1 = &Xs[(16 + n) * HDIM];
  for (int k0 = 0; k0 < HDIM; k0 += 32) {
    v16bf a0 = a_frag(arow0, k0, h);
    v16bf a1 = a_frag(arow1, k0, h);
#pragma unroll
    for (int t = 0; t < 6; ++t) {
      v16bf b = *(const v16bf*)(W1t + (size_t)((nbase + t) * 16 + n) * HDIM
                                + k0 + h * 16);
      acc[0][t] = wmma_bf16(a0, b, acc[0][t]);
      acc[1][t] = wmma_bf16(a1, b, acc[1][t]);
    }
  }
  // ReLU, write intermediate (row = mt*16 + h*8 + v, col n0+n).
#pragma unroll
  for (int mt = 0; mt < 2; ++mt)
#pragma unroll
    for (int t = 0; t < 6; ++t) {
      const int n0 = (nbase + t) * 16;
#pragma unroll
      for (int v = 0; v < 8; ++v)
        Ts[(size_t)(mt * 16 + h * 8 + v) * HDIM + n0 + n] =
            (bf16_t)fmaxf(acc[mt][t][v], 0.0f);
    }
  if (tid < 32) rowsq[tid] = 0.0f;
  __syncthreads();

  // ---- Layer 2: each wave owns one 16-col N-tile of E=128 ----
  v8f acc2[2];
  {
    const float bv = b2[wave * 16 + n];
#pragma unroll
    for (int v = 0; v < 8; ++v) { acc2[0][v] = bv; acc2[1][v] = bv; }
  }
  const bf16_t* trow0 = &Ts[n * HDIM];
  const bf16_t* trow1 = &Ts[(16 + n) * HDIM];
  for (int k0 = 0; k0 < HDIM; k0 += 32) {
    v16bf a0 = a_frag(trow0, k0, h);
    v16bf a1 = a_frag(trow1, k0, h);
    v16bf b = *(const v16bf*)(W2t + (size_t)(wave * 16 + n) * HDIM + k0 + h * 16);
    acc2[0] = wmma_bf16(a0, b, acc2[0]);
    acc2[1] = wmma_bf16(a1, b, acc2[1]);
  }

  // Mask per token row; per-row sum of squares across all 128 cols.
#pragma unroll
  for (int mt = 0; mt < 2; ++mt)
#pragma unroll
    for (int v = 0; v < 8; ++v) {
      const int m = mt * 16 + h * 8 + v;
      const float mk = (float)mask[tb + m];
      acc2[mt][v] *= mk;
      atomicAdd(&rowsq[m], acc2[mt][v] * acc2[mt][v]);
    }
  __syncthreads();

  // L2-normalize (x / max(||x||, 1e-12)) and store bf16.
#pragma unroll
  for (int mt = 0; mt < 2; ++mt)
#pragma unroll
    for (int v = 0; v < 8; ++v) {
      const int m = mt * 16 + h * 8 + v;
      const float inv = 1.0f / fmaxf(sqrtf(rowsq[m]), 1e-12f);
      out[(size_t)(tb + m) * EDIM + wave * 16 + n] = (bf16_t)(acc2[mt][v] * inv);
    }
}

// ---- Kernel 3: cartesian MaxSim -------------------------------------------
// One wave per (doc d, query pair {q0,q0+1}). Each de B-fragment feeds
// 2 queries x 2 M-tiles = 4 WMMAs (halves de L2 traffic vs 1 query/block).
__global__ __launch_bounds__(32) void maxsim_kernel(
    const bf16_t* __restrict__ qe,      // [64*32, 128] bf16
    const bf16_t* __restrict__ de,      // [256*180, 128] bf16
    const int*    __restrict__ qmask,   // [64, 32]
    float* __restrict__ out)            // [64, 256]
{
  const int d  = blockIdx.x;
  const int q0 = blockIdx.y * 2;
  const int lane = threadIdx.x;
  const int h = lane >> 4, n = lane & 15;

  // query lengths (sum of qmask row), wave reduce.
  float qlen[2];
#pragma unroll
  for (int g = 0; g < 2; ++g) {
    float v = (float)qmask[(q0 + g) * LQ + lane];
#pragma unroll
    for (int m = 1; m < 32; m <<= 1) v += __shfl_xor(v, m, 32);
    qlen[g] = v;
  }

  // Preload all A fragments (2 queries x 2 M-tiles x 4 K-chunks).
  v16bf afr[2][2][4];
#pragma unroll
  for (int g = 0; g < 2; ++g)
#pragma unroll
    for (int mt = 0; mt < 2; ++mt) {
      const bf16_t* row = qe + ((size_t)(q0 + g) * LQ + mt * 16 + n) * EDIM;
#pragma unroll
      for (int kc = 0; kc < 4; ++kc) afr[g][mt][kc] = a_frag(row, kc * 32, h);
    }

  float rmax[2][2][8];
#pragma unroll
  for (int g = 0; g < 2; ++g)
#pragma unroll
    for (int mt = 0; mt < 2; ++mt)
#pragma unroll
      for (int v = 0; v < 8; ++v) rmax[g][mt][v] = -3.0e38f;

  for (int nt = 0; nt < 12; ++nt) {        // 12*16 = 192 >= 180 doc tokens
    const int j = nt * 16 + n;             // this lane's doc-token column
    const bool valid = (j < LD);
    v16bf bfr[4];
    if (valid) {
      const bf16_t* drow = de + ((size_t)d * LD + j) * EDIM;
#pragma unroll
      for (int kc = 0; kc < 4; ++kc)
        bfr[kc] = *(const v16bf*)(drow + kc * 32 + h * 16);
    } else {
#pragma unroll
      for (int kc = 0; kc < 4; ++kc) bfr[kc] = zero16();
    }
#pragma unroll
    for (int g = 0; g < 2; ++g)
#pragma unroll
      for (int mt = 0; mt < 2; ++mt) {
        v8f c = {};
#pragma unroll
        for (int kc = 0; kc < 4; ++kc) c = wmma_bf16(afr[g][mt][kc], bfr[kc], c);
        if (valid) {
#pragma unroll
          for (int v = 0; v < 8; ++v)
            rmax[g][mt][v] = fmaxf(rmax[g][mt][v], c[v]);
        }
      }
  }

  // Max across the 16 lanes holding different columns of the same rows.
#pragma unroll
  for (int g = 0; g < 2; ++g)
#pragma unroll
    for (int mt = 0; mt < 2; ++mt)
#pragma unroll
      for (int v = 0; v < 8; ++v) {
        float r = rmax[g][mt][v];
        r = fmaxf(r, __shfl_xor(r, 1, 32));
        r = fmaxf(r, __shfl_xor(r, 2, 32));
        r = fmaxf(r, __shfl_xor(r, 4, 32));
        r = fmaxf(r, __shfl_xor(r, 8, 32));
        rmax[g][mt][v] = r;
      }

  // Sum over query-token rows (lane half h covers rows h*8..h*8+7 per tile).
#pragma unroll
  for (int g = 0; g < 2; ++g) {
    float s = 0.0f;
#pragma unroll
    for (int mt = 0; mt < 2; ++mt)
#pragma unroll
      for (int v = 0; v < 8; ++v) s += rmax[g][mt][v];
    const float tot = __shfl(s, 0, 32) + __shfl(s, 16, 32);
    if (lane == 0) out[(size_t)(q0 + g) * ND + d] = tot / qlen[g];
  }
}

// ---- Launch ---------------------------------------------------------------

extern "C" void kernel_launch(void* const* d_in, const int* in_sizes, int n_in,
                              void* d_out, int out_size, void* d_ws, size_t ws_size,
                              hipStream_t stream) {
  const float* q_hidden = (const float*)d_in[0];  // [64,32,768]
  const float* d_hidden = (const float*)d_in[1];  // [256,180,768]
  const int*   q_mask   = (const int*)d_in[2];    // [64,32]
  const int*   d_mask   = (const int*)d_in[3];    // [256,180]
  const float* W1 = (const float*)d_in[4];        // [768,768]
  const float* b1 = (const float*)d_in[5];        // [768]
  const float* W2 = (const float*)d_in[6];        // [768,128]
  const float* b2 = (const float*)d_in[7];        // [128]
  float* out = (float*)d_out;                     // [64,256]

  // Workspace layout (bf16 regions, all 256B aligned).
  char* ws = (char*)d_ws;
  bf16_t* W1t = (bf16_t*)(ws);                                   // 1,179,648 B
  bf16_t* W2t = (bf16_t*)(ws + 1179648);                         //   196,608 B
  bf16_t* qe  = (bf16_t*)(ws + 1179648 + 196608);                //   524,288 B
  bf16_t* de  = (bf16_t*)(ws + 1179648 + 196608 + 524288);       // 11,796,480 B

  const int convN = HDIM * HDIM + HDIM * EDIM;
  convert_weights<<<(convN + 255) / 256, 256, 0, stream>>>(W1, W2, W1t, W2t);

  proj_kernel<<<(NQ * LQ) / 32, 256, 0, stream>>>(
      q_hidden, q_mask, W1t, b1, W2t, b2, qe);
  proj_kernel<<<(ND * LD) / 32, 256, 0, stream>>>(
      d_hidden, d_mask, W1t, b1, W2t, b2, de);

  maxsim_kernel<<<dim3(ND, NQ / 2), 32, 0, stream>>>(qe, de, q_mask, out);
}
